// MAGNET_16604343566508
// MI455X (gfx1250) — matrix operand order repaired and checked by reading, not compiled, not run on hardware
//
#include <hip/hip_runtime.h>
#include <hip/hip_bf16.h>
#include <math.h>

// ---------------- problem constants ----------------
#define BB   8
#define NTOT 192
#define NB   64
#define DD   256
#define LDSTRIDE 200   // padded bf16 row stride for the k_ds LDS diff tile

typedef __attribute__((ext_vector_type(16))) __bf16 v16bf;
typedef __attribute__((ext_vector_type(8)))  float  v8f;

__device__ __forceinline__ float sigmoidf_(float x){ return 1.0f/(1.0f+__expf(-x)); }

__device__ __forceinline__ v8f wmma_bf16(v16bf a, v16bf b, v8f c){
  return __builtin_amdgcn_wmma_f32_16x16x32_bf16(false, a, false, b, (short)0, c, false, false);
}

// |u-v| without the fabs integer-mask idiom (works around gfx1250 ISel crash
// on v2f32-packed fabs combines).
#define ABSD_BF16(u,v) ((__bf16)fmaxf((u)-(v), (v)-(u)))

// A fragment (16x32 bf16): row-major fp32 source. lane&15 = m; element t -> k =
// t + (lane>=16?16:0). 16 contiguous floats -> float4 loads.
__device__ __forceinline__ v16bf ldA(const float* row, int k0, int lane, float scale){
  const int koff = (lane & 16) ? 16 : 0;
  const float4 x0 = *(const float4*)(row + k0 + koff);
  const float4 x1 = *(const float4*)(row + k0 + koff + 4);
  const float4 y0 = *(const float4*)(row + k0 + koff + 8);
  const float4 y1 = *(const float4*)(row + k0 + koff + 12);
  v16bf a;
  a[0]=(__bf16)(x0.x*scale); a[1]=(__bf16)(x0.y*scale); a[2]=(__bf16)(x0.z*scale); a[3]=(__bf16)(x0.w*scale);
  a[4]=(__bf16)(x1.x*scale); a[5]=(__bf16)(x1.y*scale); a[6]=(__bf16)(x1.z*scale); a[7]=(__bf16)(x1.w*scale);
  a[8]=(__bf16)(y0.x*scale); a[9]=(__bf16)(y0.y*scale); a[10]=(__bf16)(y0.z*scale); a[11]=(__bf16)(y0.w*scale);
  a[12]=(__bf16)(y1.x*scale); a[13]=(__bf16)(y1.y*scale); a[14]=(__bf16)(y1.z*scale); a[15]=(__bf16)(y1.w*scale);
  return a;
}

// B fragment (32x16 bf16): B[k][n] with k contiguous at fixed column pointer
// (weight rows w[n][k] or transposed activations). lane&15 = n; element t ->
// k = k0 + t + (lane>=16?16:0). 16 contiguous floats.
__device__ __forceinline__ v16bf ldBcol(const float* col, int k0, int lane, float scale){
  const int koff = (lane & 16) ? 16 : 0;
  const float4 x0 = *(const float4*)(col + k0 + koff);
  const float4 x1 = *(const float4*)(col + k0 + koff + 4);
  const float4 x2 = *(const float4*)(col + k0 + koff + 8);
  const float4 x3 = *(const float4*)(col + k0 + koff + 12);
  v16bf b;
  b[0]=(__bf16)(x0.x*scale); b[1]=(__bf16)(x0.y*scale); b[2]=(__bf16)(x0.z*scale); b[3]=(__bf16)(x0.w*scale);
  b[4]=(__bf16)(x1.x*scale); b[5]=(__bf16)(x1.y*scale); b[6]=(__bf16)(x1.z*scale); b[7]=(__bf16)(x1.w*scale);
  b[8]=(__bf16)(x2.x*scale); b[9]=(__bf16)(x2.y*scale); b[10]=(__bf16)(x2.z*scale); b[11]=(__bf16)(x2.w*scale);
  b[12]=(__bf16)(x3.x*scale); b[13]=(__bf16)(x3.y*scale); b[14]=(__bf16)(x3.z*scale); b[15]=(__bf16)(x3.w*scale);
  return b;
}

// B fragment gathered from a row-major matrix M[k][n] (stride ld over k).
__device__ __forceinline__ v16bf ldBgather(const float* base, int ld, int n, int k0, int lane){
  const int koff = (lane & 16) ? 16 : 0;
  v16bf b;
#pragma unroll
  for(int t=0;t<16;t++) b[t] = (__bf16)base[(size_t)(k0+koff+t)*ld + n];
  return b;
}

// Same, with associated-matrix transform ae = (x>0 ? 1 : x).
__device__ __forceinline__ v16bf ldBgatherAe(const float* base, int ld, int n, int k0, int lane){
  const int koff = (lane & 16) ? 16 : 0;
  v16bf b;
#pragma unroll
  for(int t=0;t<16;t++){
    float v = base[(size_t)(k0+koff+t)*ld + n];
    v = (v > 0.0f) ? 1.0f : v;
    b[t] = (__bf16)v;
  }
  return b;
}

// ---------------- 1) projection: relu(LN(x @ w.T + b)) ----------------
__global__ void __launch_bounds__(512) k_proj(const float* __restrict__ X, const float* __restrict__ W,
                                              const float* __restrict__ bias, const float* __restrict__ lng,
                                              const float* __restrict__ lnb, float* __restrict__ fn,
                                              int Kin, int blk_off){
  __shared__ float tile[16][264];
  const int tid = threadIdx.x, lane = tid & 31, wv = tid >> 5;  // 16 waves
  const int rt = blockIdx.x;                                    // rows rt*16..+15 (of 512)
  const int n = wv*16 + (lane & 15);
  {
    const int rg = rt*16 + (lane & 15);
    const float* arow = X + (size_t)rg * Kin;
    const float* bcol = W + (size_t)n * Kin;
    v8f c = {};
#pragma unroll 2
    for(int k0=0;k0<Kin;k0+=32)
      c = wmma_bf16(ldA(arow,k0,lane,1.0f), ldBcol(bcol,k0,lane,1.0f), c);
    const int mo = (lane & 16) ? 8 : 0;
#pragma unroll
    for(int r=0;r<8;r++) tile[r+mo][n] = c[r] + bias[n];
  }
  __syncthreads();
  {
    const int row = wv;
    float vals[8], s=0.f, ss=0.f;
#pragma unroll
    for(int q=0;q<8;q++){ float v = tile[row][lane+32*q]; vals[q]=v; s+=v; ss+=v*v; }
    for(int m=16;m>=1;m>>=1){ s += __shfl_xor(s,m,32); ss += __shfl_xor(ss,m,32); }
    const float mean = s*(1.0f/256.0f);
    const float var  = ss*(1.0f/256.0f) - mean*mean;
    const float inv  = rsqrtf(var + 1e-5f);
    const int rg = rt*16 + row, b = rg/64, nloc = rg%64;
    float* dst = fn + ((size_t)(b*NTOT + blk_off + nloc))*DD;
#pragma unroll
    for(int q=0;q<8;q++){
      int d = lane + 32*q;
      float o = (vals[q]-mean)*inv*lng[d] + lnb[d];
      dst[d] = fmaxf(o, 0.0f);
    }
  }
}

// ---------------- row inverse-L2-norms of fn ----------------
__global__ void __launch_bounds__(256) k_rnorm(const float* __restrict__ fn, float* __restrict__ rn){
  const int tid=threadIdx.x, lane=tid&31, wv=tid>>5;
  const int row = blockIdx.x*8 + wv;  // grid 192 -> 1536 rows
  const float* p = fn + (size_t)row*DD;
  float ss=0.f;
#pragma unroll
  for(int q=0;q<8;q++){ float v = p[lane+32*q]; ss += v*v; }
  for(int m=16;m>=1;m>>=1) ss += __shfl_xor(ss,m,32);
  if(lane==0) rn[row] = 1.0f / fmaxf(sqrtf(ss), 1e-12f);
}

// ---------------- 2) block cosine sims: ep = relu(v v^T - 1e-4) ----------------
__global__ void __launch_bounds__(32) k_cossim(const float* __restrict__ fn, const float* __restrict__ rn,
                                               float* __restrict__ outfeat, float* __restrict__ fe){
  const int lane = threadIdx.x;
  const int bid = blockIdx.x;              // b*48 + blk*16 + it*4 + jt
  const int jt = bid & 3, it = (bid>>2) & 3, blk = (bid/16)%3, b = bid/48;
  const int base = b*NTOT + blk*NB;
  const int mi = base + it*16 + (lane & 15);
  const int nj = base + jt*16 + (lane & 15);
  const float sa = rn[mi], sb = rn[nj];
  const float* arow = fn + (size_t)mi*DD;
  const float* bcol = fn + (size_t)nj*DD;
  v8f c = {};
#pragma unroll 2
  for(int k0=0;k0<DD;k0+=32)
    c = wmma_bf16(ldA(arow,k0,lane,sa), ldBcol(bcol,k0,lane,sb), c);
  const int mo = (lane & 16) ? 8 : 0;
  const int jg = blk*NB + jt*16 + (lane & 15);
#pragma unroll
  for(int r=0;r<8;r++){
    const int ig = blk*NB + it*16 + r + mo;
    const float ep = fmaxf(c[r] - 1e-4f, 0.0f);
    const size_t o = ((size_t)b*NTOT + ig)*NTOT + jg;
    outfeat[o] = ep;
    fe[o] = ep;
  }
}

// ---------------- 3) d12 = deg^{-1/2} ----------------
__global__ void __launch_bounds__(256) k_d12(const float* __restrict__ fe, float* __restrict__ d12){
  const int tid=threadIdx.x, lane=tid&31, wv=tid>>5;
  const int row = blockIdx.x*8 + wv;
  const float* p = fe + (size_t)row*NTOT;
  float s=0.f;
#pragma unroll
  for(int q=0;q<6;q++) s += p[lane+32*q];
  for(int m=16;m>=1;m>>=1) s += __shfl_xor(s,m,32);
  if(lane==0) d12[row] = rsqrtf(s);
}

// ---------------- 4) S = fe * d12_i * d12_j ----------------
__global__ void __launch_bounds__(256) k_S(const float* __restrict__ fe, const float* __restrict__ d12,
                                           float* __restrict__ S){
  const int idx = blockIdx.x*256 + threadIdx.x;   // grid 1152, 294912 elems
  const int j = idx % NTOT, bi = idx / NTOT, b = bi / NTOT;
  S[idx] = fe[idx] * d12[bi] * d12[b*NTOT + j];
}

// ---------------- 5) T1 = S @ ae ----------------
__global__ void __launch_bounds__(32) k_T1(const float* __restrict__ S, const float* __restrict__ am,
                                           float* __restrict__ T1){
  const int lane=threadIdx.x, bid=blockIdx.x;     // b*144 + it*12 + jt
  const int jt=bid%12, it=(bid/12)%12, b=bid/144;
  const float* arow  = S + ((size_t)b*NTOT + it*16 + (lane&15))*NTOT;
  const float* bbase = am + (size_t)b*NTOT*NTOT;
  const int n = jt*16 + (lane&15);
  v8f c = {};
#pragma unroll 2
  for(int k0=0;k0<NTOT;k0+=32)
    c = wmma_bf16(ldA(arow,k0,lane,1.0f), ldBgatherAe(bbase,NTOT,n,k0,lane), c);
  const int mo = (lane&16)?8:0;
#pragma unroll
  for(int r=0;r<8;r++)
    T1[((size_t)b*NTOT + it*16 + r + mo)*NTOT + n] = c[r];
}

// ---------------- 6) C = T1 @ S^T ----------------
__global__ void __launch_bounds__(32) k_C(const float* __restrict__ T1, const float* __restrict__ S,
                                          float* __restrict__ C){
  const int lane=threadIdx.x, bid=blockIdx.x;
  const int jt=bid%12, it=(bid/12)%12, b=bid/144;
  const float* arow = T1 + ((size_t)b*NTOT + it*16 + (lane&15))*NTOT;
  const int n = jt*16 + (lane&15);
  const float* bcol = S + ((size_t)b*NTOT + n)*NTOT;   // B[k][n] = S[b][n][k]
  v8f c = {};
#pragma unroll 2
  for(int k0=0;k0<NTOT;k0+=32)
    c = wmma_bf16(ldA(arow,k0,lane,1.0f), ldBcol(bcol,k0,lane,1.0f), c);
  const int mo = (lane&16)?8:0;
#pragma unroll
  for(int r=0;r<8;r++)
    C[((size_t)b*NTOT + it*16 + r + mo)*NTOT + n] = c[r];
}

// ---------------- 7) alpha gate + mix: fe = al*C + (1-al)*fe ----------------
__global__ void __launch_bounds__(192) k_alpha(const float* __restrict__ C, const float* __restrict__ aw,
                                               const float* __restrict__ ab, float* __restrict__ fe){
  __shared__ float red[192];
  const int j = threadIdx.x, bi = blockIdx.x;     // bi = b*192+i
  const size_t o = (size_t)bi*NTOT + j;
  const float f = fe[o], cc = C[o];
  red[j] = f*aw[j] + cc*aw[NTOT+j];
  __syncthreads();
  if(j<64) red[j] += red[j+64] + red[j+128];
  __syncthreads();
  if(j<32) red[j] += red[j+32];
  __syncthreads();
  if(j<16) red[j] += red[j+16];
  __syncthreads();
  if(j<8)  red[j] += red[j+8];
  __syncthreads();
  if(j<4)  red[j] += red[j+4];
  __syncthreads();
  if(j<2)  red[j] += red[j+2];
  __syncthreads();
  if(j==0) red[0] += red[1];
  __syncthreads();
  const float al = sigmoidf_(red[0] + ab[0]);
  fe[o] = al*cc + (1.0f-al)*f;
}

// ---------------- 8) dn = leaky_relu([fe,dn] @ p2d_w.T + b, 0.01) ----------------
__global__ void __launch_bounds__(32) k_p2d(const float* __restrict__ fe, const float* __restrict__ dnsrc,
                                            const float* __restrict__ W, const float* __restrict__ bias,
                                            float* __restrict__ dndst){
  const int lane=threadIdx.x, bid=blockIdx.x;     // rt*12 + nt, rt in 0..95
  const int nt = bid%12, rt = bid/12;
  const int rg = rt*16 + (lane&15);
  const int b = rg / NTOT, i = rg % NTOT;
  const float* feR = fe    + ((size_t)b*NTOT + i)*NTOT;
  const float* dnR = dnsrc + ((size_t)b*NTOT + i)*NTOT;
  const int n = nt*16 + (lane&15);
  const float* bcol = W + (size_t)n*(2*NTOT);
  v8f c = {};
  const int koff = (lane&16)?16:0;
#pragma unroll 2
  for(int k0=0;k0<2*NTOT;k0+=32){
    const int kr0 = k0+koff, kr1 = k0+koff+8;
    const float* r0 = (kr0 < NTOT) ? (feR + kr0) : (dnR + kr0 - NTOT);
    const float* r1 = (kr1 < NTOT) ? (feR + kr1) : (dnR + kr1 - NTOT);
    const float4 x0 = *(const float4*)r0, x1 = *(const float4*)(r0+4);
    const float4 y0 = *(const float4*)r1, y1 = *(const float4*)(r1+4);
    v16bf a;
    a[0]=(__bf16)x0.x; a[1]=(__bf16)x0.y; a[2]=(__bf16)x0.z; a[3]=(__bf16)x0.w;
    a[4]=(__bf16)x1.x; a[5]=(__bf16)x1.y; a[6]=(__bf16)x1.z; a[7]=(__bf16)x1.w;
    a[8]=(__bf16)y0.x; a[9]=(__bf16)y0.y; a[10]=(__bf16)y0.z; a[11]=(__bf16)y0.w;
    a[12]=(__bf16)y1.x; a[13]=(__bf16)y1.y; a[14]=(__bf16)y1.z; a[15]=(__bf16)y1.w;
    c = wmma_bf16(a, ldBcol(bcol,k0,lane,1.0f), c);
  }
  const int mo = (lane&16)?8:0;
#pragma unroll
  for(int r=0;r<8;r++){
    const int rg2 = rt*16 + r + mo;
    float v = c[r] + bias[n];
    v = (v >= 0.0f) ? v : 0.01f*v;
    dndst[((size_t)(rg2/NTOT)*NTOT + (rg2%NTOT))*NTOT + n] = v;
  }
}

// ---------------- 9a) repack ds_w1 (fp32 [384,192]) into bf16 fragment layout ----
// out[((nt*6+ks)*32+lane)*16 + t] = (bf16) W1[nt*16 + (lane&15)][ks*32 + ((lane&16)?16:0) + t]
__global__ void __launch_bounds__(32) k_packw1(const float* __restrict__ W1, __bf16* __restrict__ out){
  const int lane = threadIdx.x, bid = blockIdx.x;   // bid = nt*6 + ks, grid 144
  const int ks = bid % 6, nt = bid / 6;
  const int n  = nt*16 + (lane & 15);
  const int kb = ks*32 + ((lane & 16) ? 16 : 0);
  const float* src = W1 + (size_t)n*NTOT + kb;
  __bf16* dst = out + ((size_t)bid*32 + lane)*16;
#pragma unroll
  for(int t=0;t<16;t++) dst[t] = (__bf16)src[t];
}

// ---------------- 9b) fused DistributionSimilarity ----------------
// Per workgroup: (b, i, jt). The 16x192 bf16 |dn_i - dn_j| tile is built ONCE
// in LDS (shared by all 24 waves); B fragments come pre-packed in bf16.
// Inner iteration: 2 LDS b128 + 2 global b128 + 1 v_wmma.
__global__ void __launch_bounds__(768) k_ds(const float* __restrict__ dn, const __bf16* __restrict__ W1p,
                                            const float* __restrict__ b1, const float* __restrict__ prelu,
                                            const float* __restrict__ W2, const float* __restrict__ b2,
                                            float* __restrict__ ed, float* __restrict__ outdist){
  __shared__ __align__(16) __bf16 atile[16*LDSTRIDE];
  __shared__ float acc[16];
  const int tid=threadIdx.x, lane=tid&31, wv=tid>>5;    // 24 waves
  const int bid=blockIdx.x;                             // b*(192*12) + i*12 + jt
  const int jt = bid%12, i = (bid/12)%NTOT, b = bid/(12*NTOT);
  if(tid<16) acc[tid]=0.0f;
  // cooperative fill: 16 rows x 192 k = 3072 bf16, 4 per thread
  {
    const int row = tid/48, kq = (tid%48)*4;
    const float4 di = *(const float4*)(dn + ((size_t)b*NTOT + i)*NTOT + kq);
    const float4 dj = *(const float4*)(dn + ((size_t)b*NTOT + jt*16 + row)*NTOT + kq);
    __bf16* p = atile + row*LDSTRIDE + kq;
    p[0] = ABSD_BF16(di.x, dj.x);
    p[1] = ABSD_BF16(di.y, dj.y);
    p[2] = ABSD_BF16(di.z, dj.z);
    p[3] = ABSD_BF16(di.w, dj.w);
  }
  __syncthreads();
  const int n = wv*16 + (lane&15);
  const int koff = (lane&16)?16:0;
  const __bf16* arow = atile + (lane&15)*LDSTRIDE + koff;
  const uint4* wfrag = (const uint4*)W1p + (((size_t)wv*6)*32 + lane)*2;
  v8f c = {};
#pragma unroll
  for(int ks=0;ks<6;ks++){
    union { v16bf v; uint4 q[2]; } a, bb;
    a.q[0] = *(const uint4*)(arow + ks*32);
    a.q[1] = *(const uint4*)(arow + ks*32 + 8);
    bb.q[0] = wfrag[(size_t)ks*64 + 0];
    bb.q[1] = wfrag[(size_t)ks*64 + 1];
    c = wmma_bf16(a.v, bb.v, c);
  }
  const float p = prelu[0];
  const float w2n = W2[n];
  const float b1n = b1[n];
  const int mo = (lane&16)?8:0;
#pragma unroll
  for(int r=0;r<8;r++){
    float h = c[r] + b1n;
    h = (h >= 0.0f) ? h : p*h;
    atomicAdd(&acc[r+mo], h*w2n);
  }
  __syncthreads();
  if(tid<16){
    const float e = sigmoidf_(acc[tid] + b2[0]);
    const size_t o = ((size_t)b*NTOT + i)*NTOT + jt*16 + tid;
    ed[o] = e;
    outdist[o] = e;
  }
}

// ---------------- 10) ef = l1norm over block with diag zeroed ----------------
__global__ void __launch_bounds__(64) k_ef(const float* __restrict__ ed, float* __restrict__ ef){
  __shared__ float red[64];
  const int j = threadIdx.x, bi = blockIdx.x;     // bi = b*192+i
  const int i = bi % NTOT, blk = i / NB, iloc = i % NB;
  const float v = (j==iloc) ? 0.0f : (ed[(size_t)bi*NTOT + blk*NB + j] + 1e-6f);
  red[j] = v;
  __syncthreads();
  if(j<32) red[j]+=red[j+32];
  __syncthreads();
  if(j<16) red[j]+=red[j+16];
  __syncthreads();
  if(j<8)  red[j]+=red[j+8];
  __syncthreads();
  if(j<4)  red[j]+=red[j+4];
  __syncthreads();
  if(j<2)  red[j]+=red[j+2];
  __syncthreads();
  if(j==0) red[0]+=red[1];
  __syncthreads();
  const float s = fmaxf(red[0], 1e-12f);
  ef[(size_t)bi*NB + j] = v / s;
}

// ---------------- 11a) tagg = ef @ fn_block ----------------
__global__ void __launch_bounds__(32) k_tagg(const float* __restrict__ ef, const float* __restrict__ fn,
                                             float* __restrict__ tagg){
  const int lane=threadIdx.x, bid=blockIdx.x;     // grid 1536
  const int dt=bid%16, it=(bid/16)%4, blk=(bid/64)%3, b=bid/192;
  const int iglob = blk*NB + it*16 + (lane&15);
  const float* arow  = ef + ((size_t)b*NTOT + iglob)*NB;
  const float* bbase = fn + ((size_t)b*NTOT + blk*NB)*DD;  // B[k][n] strided gather
  const int n = dt*16 + (lane&15);
  v8f c = {};
#pragma unroll
  for(int k0=0;k0<NB;k0+=32)
    c = wmma_bf16(ldA(arow,k0,lane,1.0f), ldBgather(bbase,DD,n,k0,lane), c);
  const int mo = (lane&16)?8:0;
#pragma unroll
  for(int r=0;r<8;r++)
    tagg[((size_t)b*NTOT + blk*NB + it*16 + r + mo)*DD + n] = c[r];
}

// ---------------- 11b) fused D2FAgg: fi, aggr, beta, mix, LN, relu ----------------
__global__ void __launch_bounds__(512) k_d2f(const float* __restrict__ fn, const float* __restrict__ tagg,
                                             const float* __restrict__ featW, const float* __restrict__ featB,
                                             const float* __restrict__ rawW, const float* __restrict__ rawB,
                                             const float* __restrict__ betaW, const float* __restrict__ betaB,
                                             const float* __restrict__ lnG, const float* __restrict__ lnB,
                                             float* __restrict__ fndst, int blk){
  __shared__ float fiL[16][264];
  __shared__ float agL[16][264];
  const int tid=threadIdx.x, lane=tid&31, wv=tid>>5;   // 16 waves
  const int rt = blockIdx.x;                           // 0..31, 512 block-local rows
  const int n = wv*16 + (lane&15);
  {
    const int r = rt*16 + (lane&15);
    const int b = r/NB, iloc = r%NB;
    const size_t rowoff = ((size_t)b*NTOT + blk*NB + iloc)*DD;
    const float* aF = fn   + rowoff;
    const float* aG = tagg + rowoff;
    const float* bF = featW + (size_t)n*DD;
    const float* bR = rawW  + (size_t)n*DD;
    v8f cf = {}, ca = {};
#pragma unroll 2
    for(int k0=0;k0<DD;k0+=32){
      cf = wmma_bf16(ldA(aF,k0,lane,1.0f), ldBcol(bF,k0,lane,1.0f), cf);
      ca = wmma_bf16(ldA(aG,k0,lane,1.0f), ldBcol(bR,k0,lane,1.0f), ca);
    }
    const int mo = (lane&16)?8:0;
#pragma unroll
    for(int r2=0;r2<8;r2++){
      fiL[r2+mo][n] = cf[r2] + featB[n];
      agL[r2+mo][n] = ca[r2] + rawB[n];
    }
  }
  __syncthreads();
  {
    const int row = wv;
    float fiv[8], agv[8], part=0.f;
#pragma unroll
    for(int q=0;q<8;q++){
      const int d = lane + 32*q;
      const float fi = fiL[row][d], ag = agL[row][d];
      fiv[q]=fi; agv[q]=ag;
      part += fi*betaW[d] + ag*betaW[DD+d] + (fi-ag)*betaW[2*DD+d];
    }
    for(int m=16;m>=1;m>>=1) part += __shfl_xor(part,m,32);
    const float beta = sigmoidf_(part + betaB[0]);
    float mx[8], s=0.f, ss=0.f;
#pragma unroll
    for(int q=0;q<8;q++){
      const float v = fiv[q]*beta + agv[q]*(1.0f-beta);
      mx[q]=v; s+=v; ss+=v*v;
    }
    for(int m=16;m>=1;m>>=1){ s += __shfl_xor(s,m,32); ss += __shfl_xor(ss,m,32); }
    const float mean = s*(1.0f/256.0f);
    const float var  = ss*(1.0f/256.0f) - mean*mean;
    const float inv  = rsqrtf(var + 1e-5f);
    const int r = rt*16 + row, b = r/NB, iloc = r%NB;
    float* dst = fndst + ((size_t)b*NTOT + blk*NB + iloc)*DD;
#pragma unroll
    for(int q=0;q<8;q++){
      const int d = lane + 32*q;
      const float o = (mx[q]-mean)*inv*lnG[d] + lnB[d];
      dst[d] = fmaxf(o, 0.0f);
    }
  }
}

// ---------------- host side ----------------
// Param leaves in jax pytree (sorted-key) order; element counts per leaf.
static const int kParamSizes[78] = {
  256,256,256,131072,                    // body_b, body_ln_b, body_ln_g, body_w
  256,256,256,131072,                    // face_*
  // g0:
  1,384, 384,1,1,73728,384,              // alpha_b, alpha_w, ds_b1, ds_b2, ds_prelu, ds_w1, ds_w2
  1,768,256,65536,256,256,256,65536,     // m0: beta_b,beta_w,feat_b,feat_w,ln_b,ln_g,raw_b,raw_w
  1,768,256,65536,256,256,256,65536,     // m1
  1,768,256,65536,256,256,256,65536,     // m2
  192,73728,                             // p2d_b, p2d_w
  // g1:
  1,384, 384,1,1,73728,384,
  1,768,256,65536,256,256,256,65536,
  1,768,256,65536,256,256,256,65536,
  1,768,256,65536,256,256,256,65536,
  192,73728,
  256,256,256,262144                     // voice_b, voice_ln_b, voice_ln_g, voice_w
};

extern "C" void kernel_launch(void* const* d_in, const int* in_sizes, int n_in,
                              void* d_out, int out_size, void* d_ws, size_t ws_size,
                              hipStream_t stream){
  (void)in_sizes; (void)ws_size;
  // resolve param pointers (separate leaves, or single concatenated blob fallback)
  const float* pp[78];
  if(n_in >= 6+78){
    for(int i=0;i<78;i++) pp[i] = (const float*)d_in[6+i];
  } else {
    const float* blob = (const float*)d_in[6];
    size_t off=0;
    for(int i=0;i<78;i++){ pp[i]=blob+off; off += (size_t)kParamSizes[i]; }
  }
  const float* face  = (const float*)d_in[0];
  const float* body  = (const float*)d_in[1];
  const float* voice = (const float*)d_in[2];
  const float* dn_in = (const float*)d_in[3];
  const float* am    = (const float*)d_in[4];

  // workspace layout (floats)
  float* ws  = (float*)d_ws;
  float* fnA = ws;                    // 393216
  float* fnB = fnA + 393216;          // 393216
  float* fe  = fnB + 393216;          // 294912
  float* Sb  = fe  + 294912;
  float* T1  = Sb  + 294912;
  float* Cb  = T1  + 294912;
  float* dnA = Cb  + 294912;
  float* dnB = dnA + 294912;
  float* ed  = dnB + 294912;
  float* efb = ed  + 294912;          // 98304
  float* tg  = efb + 98304;           // 393216
  float* rn  = tg  + 393216;          // 1536
  float* d12 = rn  + 1536;            // 1536
  __bf16* w1p = (__bf16*)(d12 + 1536);// 73728 bf16 (36864 floats)

  (void)hipMemsetAsync(d_out, 0, (size_t)out_size*sizeof(float), stream);

  // projections -> fnA
  k_proj<<<32, 512, 0, stream>>>(face,  pp[7],  pp[4],  pp[6],  pp[5],  fnA,  512,   0);
  k_proj<<<32, 512, 0, stream>>>(body,  pp[3],  pp[0],  pp[2],  pp[1],  fnA,  512,  64);
  k_proj<<<32, 512, 0, stream>>>(voice, pp[77], pp[74], pp[76], pp[75], fnA, 1024, 128);

  for(int l=0;l<2;l++){
    const int G = 8 + 33*l;
    const float* g_alpha_b = pp[G+0];
    const float* g_alpha_w = pp[G+1];
    const float* g_ds_b1   = pp[G+2];
    const float* g_ds_b2   = pp[G+3];
    const float* g_ds_pre  = pp[G+4];
    const float* g_ds_w1   = pp[G+5];
    const float* g_ds_w2   = pp[G+6];
    const float* g_p2d_b   = pp[G+31];
    const float* g_p2d_w   = pp[G+32];

    float* fn_src = (l==0) ? fnA : fnB;
    float* fn_dst = (l==0) ? fnB : fnA;
    const float* dn_src = (l==0) ? dn_in : dnA;
    float* dn_dst = (l==0) ? dnA : dnB;
    float* outfeat = (float*)d_out + (size_t)l      *BB*NTOT*NTOT;
    float* outdist = (float*)d_out + (size_t)(2+l)  *BB*NTOT*NTOT;

    (void)hipMemsetAsync(fe, 0, (size_t)BB*NTOT*NTOT*sizeof(float), stream);

    k_rnorm <<<192,  256, 0, stream>>>(fn_src, rn);
    k_cossim<<<384,   32, 0, stream>>>(fn_src, rn, outfeat, fe);
    k_d12   <<<192,  256, 0, stream>>>(fe, d12);
    k_S     <<<1152, 256, 0, stream>>>(fe, d12, Sb);
    k_T1    <<<1152,  32, 0, stream>>>(Sb, am, T1);
    k_C     <<<1152,  32, 0, stream>>>(T1, Sb, Cb);
    k_alpha <<<1536, 192, 0, stream>>>(Cb, g_alpha_w, g_alpha_b, fe);
    k_p2d   <<<1152,  32, 0, stream>>>(fe, dn_src, g_p2d_w, g_p2d_b, dn_dst);
    k_packw1<<<144,   32, 0, stream>>>(g_ds_w1, w1p);
    k_ds    <<<18432,768, 0, stream>>>(dn_dst, w1p, g_ds_b1, g_ds_pre, g_ds_w2, g_ds_b2,
                                       ed, outdist);
    k_ef    <<<1536,  64, 0, stream>>>(ed, efb);
    k_tagg  <<<1536,  32, 0, stream>>>(efb, fn_src, tg);
    for(int m=0;m<3;m++){
      const int M = G + 7 + 8*m;
      k_d2f<<<32, 512, 0, stream>>>(fn_src, tg,
                                    pp[M+3], pp[M+2],   // feat_w, feat_b
                                    pp[M+7], pp[M+6],   // raw_w, raw_b
                                    pp[M+1], pp[M+0],   // beta_w, beta_b
                                    pp[M+5], pp[M+4],   // ln_g, ln_b
                                    fn_dst, m);
    }
  }
}
